// WindowAttention_61168924230269
// MI455X (gfx1250) — matrix operand order repaired and checked by reading, not compiled, not run on hardware
//
#include <hip/hip_runtime.h>
#include <hip/hip_bf16.h>

typedef __attribute__((ext_vector_type(16))) _Float16 v16h;
typedef __attribute__((ext_vector_type(8)))  float    v8f;

#define DIM    192
#define NHEADS 6
#define HD     32
#define WSZ    8
#define DISP   4
#define SCALE  0.17677669529663687f

// pitches (in elements; SPP in f32); all f16 row strides are multiples of 8 halves (16B)
#define XP  200   // x window tile pitch
#define QP  40    // q/k pitch (also reused as per-head O staging pitch)
#define VP  72    // vT pitch
#define SPP 66    // S (f32) pitch
#define PP  72    // P (f16) pitch, aliased onto S buffer

struct H16pack { uint4 a, b; };

__device__ __forceinline__ v16h ld_frag(const _Float16* p0, const _Float16* p1) {
    H16pack t;
    t.a = *(const uint4*)p0;
    t.b = *(const uint4*)p1;
    return __builtin_bit_cast(v16h, t);
}

__device__ __forceinline__ v8f wmma_f16(v16h a, v16h b, v8f c) {
    return __builtin_amdgcn_wmma_f32_16x16x32_f16(false, a, false, b, (short)0, c, false, false);
}

// async global->LDS b128 copy (CDNA5 GLOBAL_LOAD_ASYNC_TO_LDS_B128, ASYNCcnt)
__device__ __forceinline__ void async_copy_b128(void* lds_dst, const void* gsrc) {
    unsigned lds_addr = (unsigned)(unsigned long long)lds_dst; // low 32 bits = LDS offset
    asm volatile("global_load_async_to_lds_b128 %0, %1, off"
                 :: "v"(lds_addr), "v"(gsrc) : "memory");
}
__device__ __forceinline__ void async_wait0() {
    asm volatile("s_wait_asynccnt 0" ::: "memory");
}

// ---------------------------------------------------------------------------
// Kernel 0: convert weights f32 -> f16 into workspace
// ---------------------------------------------------------------------------
__global__ void wa_convert_weights(const float* __restrict__ wqkv,
                                   const float* __restrict__ wout,
                                   _Float16* __restrict__ dst) {
    int i = blockIdx.x * 256 + threadIdx.x;
    if (i < 3 * DIM * DIM) {               // 110592
        dst[i] = (_Float16)wqkv[i];
    } else if (i < 3 * DIM * DIM + DIM * DIM) {
        dst[i] = (_Float16)wout[i - 3 * DIM * DIM];
    }
}

// ---------------------------------------------------------------------------
// Kernel 1: fused shifted-window QKV projection + attention.
// One block = one (batch, window). 256 threads = 8 waves.
// ---------------------------------------------------------------------------
__global__ void __launch_bounds__(256)
wa_window_attn(const float* __restrict__ x,
               const _Float16* __restrict__ wq,     // qkv weights f16 (576x192)
               const float* __restrict__ pos,       // (15,15)
               _Float16* __restrict__ attn_out)     // (262144,192) window-major
{
    __shared__ _Float16 xw[64 * XP];    // 25600 B
    __shared__ _Float16 qs[64 * QP];    //  5120 B
    __shared__ _Float16 ks[64 * QP];    //  5120 B (reused as O staging per head)
    __shared__ _Float16 vT[HD * VP];    //  4608 B
    __shared__ float    SP[64 * SPP];   // 16896 B (reused as f16 P, pitch PP)
    __shared__ float    posb[225];      //   900 B

    const int tid  = threadIdx.x;
    const int lane = tid & 31;
    const int wv   = tid >> 5;
    const int l16  = lane & 15;
    const int hi   = lane >> 4;

    const int bb  = blockIdx.x >> 10;
    const int win = blockIdx.x & 1023;
    const int wr  = win >> 5;
    const int wc  = win & 31;

    // warm the f16 weight matrix into cache (global_prefetch_b8)
    for (int off = tid * 64; off < 3 * DIM * DIM; off += 256 * 64)
        __builtin_prefetch(wq + off, 0, 3);

    // ---- load shifted x window (64 pixels x 192 ch), f32 -> f16 in LDS ----
    {
        int p    = tid >> 2;          // pixel 0..63
        int part = tid & 3;           // 48-float chunk
        int pr = p >> 3, pc = p & 7;
        int rs = ((wr << 3) + pr + DISP) & 255;
        int cs = ((wc << 3) + pc + DISP) & 255;
        const float* xrow = x + (((long)bb * 256 + rs) * 256 + cs) * DIM;
        struct H4 { _Float16 a, b, c, d; };
        #pragma unroll
        for (int k = 0; k < 12; ++k) {
            float4 f = *(const float4*)(xrow + part * 48 + k * 4);
            H4* d = (H4*)&xw[p * XP + part * 48 + k * 4];
            *d = H4{(_Float16)f.x, (_Float16)f.y, (_Float16)f.z, (_Float16)f.w};
        }
        if (tid < 225) posb[tid] = pos[tid];
    }
    __syncthreads();

    _Float16* P = (_Float16*)SP;                   // f16 alias of S buffer
    const long mbase = (long)blockIdx.x * 64;      // window-major pixel base

    for (int head = 0; head < NHEADS; ++head) {
        __syncthreads();   // previous head's O-staging reads complete

        // ---- QKV GEMM for this head: 24 tiles of 16x16, K=192 ----
        for (int t = wv; t < 24; t += 8) {
            int m0  = (t & 3) << 4;
            int nt  = t >> 2;          // 0..5
            int sel = nt >> 1;         // 0=q 1=k 2=v
            int dh  = nt & 1;          // which 16-col half of head_dim
            int o0  = sel * DIM + head * HD + dh * 16;

            v8f acc;
            #pragma unroll
            for (int z = 0; z < 8; ++z) acc[z] = 0.f;

            #pragma unroll
            for (int s = 0; s < 6; ++s) {
                const _Float16* a0 = &xw[(m0 + l16) * XP + s * 32 + hi * 8];
                v16h A = ld_frag(a0, a0 + 16);
                const _Float16* b0 = wq + (o0 + l16) * DIM + s * 32 + hi * 8;
                v16h Bf = ld_frag(b0, b0 + 16);
                acc = wmma_f16(A, Bf, acc);
            }

            if (sel == 0) {
                #pragma unroll
                for (int g = 0; g < 8; ++g)
                    qs[(m0 + hi * 8 + g) * QP + dh * 16 + l16] = (_Float16)acc[g];
            } else if (sel == 1) {
                #pragma unroll
                for (int g = 0; g < 8; ++g)
                    ks[(m0 + hi * 8 + g) * QP + dh * 16 + l16] = (_Float16)acc[g];
            } else {
                #pragma unroll
                for (int g = 0; g < 8; ++g)
                    vT[(dh * 16 + l16) * VP + m0 + hi * 8 + g] = (_Float16)acc[g];
            }
        }
        __syncthreads();

        // ---- S = q . k^T : 16 tiles of 16x16, K=32 (one WMMA each) ----
        for (int t = wv; t < 16; t += 8) {
            int ti = t >> 2, tj = t & 3;
            const _Float16* a0 = &qs[(ti * 16 + l16) * QP + hi * 8];
            v16h A = ld_frag(a0, a0 + 16);
            const _Float16* b0 = &ks[(tj * 16 + l16) * QP + hi * 8];
            v16h Bf = ld_frag(b0, b0 + 16);
            v8f acc;
            #pragma unroll
            for (int z = 0; z < 8; ++z) acc[z] = 0.f;
            acc = wmma_f16(A, Bf, acc);
            #pragma unroll
            for (int g = 0; g < 8; ++g)
                SP[(ti * 16 + hi * 8 + g) * SPP + tj * 16 + l16] = acc[g];
        }
        __syncthreads();

        // ---- softmax: 4 lanes per row, 16 cols each; rel-pos bias fused ----
        {
            int i  = tid >> 2;
            int j0 = (tid & 3) << 4;
            float vals[16];
            float vmax = -3.0e38f;
            #pragma unroll
            for (int jj = 0; jj < 16; ++jj) {
                int j = j0 + jj;
                int rh = (i >> 3) - (j >> 3) + 7;
                int rw = (i & 7) - (j & 7) + 7;
                float v = SP[i * SPP + j] * SCALE + posb[rh * 15 + rw];
                vals[jj] = v;
                vmax = fmaxf(vmax, v);
            }
            vmax = fmaxf(vmax, __shfl_xor(vmax, 1));
            vmax = fmaxf(vmax, __shfl_xor(vmax, 2));
            float sum = 0.f;
            #pragma unroll
            for (int jj = 0; jj < 16; ++jj) {
                float e = __expf(vals[jj] - vmax);
                vals[jj] = e;
                sum += e;
            }
            sum += __shfl_xor(sum, 1);
            sum += __shfl_xor(sum, 2);
            float inv = 1.0f / sum;
            __syncthreads();            // all S reads done; safe to alias as P
            #pragma unroll
            for (int jj = 0; jj < 16; ++jj)
                P[i * PP + j0 + jj] = (_Float16)(vals[jj] * inv);
        }
        __syncthreads();

        // ---- O = P . V : 8 tiles of 16x16, K=64 (one tile per wave) ----
        // Stage O in LDS (reuse ks: dead after S step), then coalesced store.
        {
            int m0 = (wv & 3) << 4;
            int dh = wv >> 2;
            v8f acc;
            #pragma unroll
            for (int z = 0; z < 8; ++z) acc[z] = 0.f;
            #pragma unroll
            for (int s = 0; s < 2; ++s) {
                const _Float16* a0 = &P[(m0 + l16) * PP + s * 32 + hi * 8];
                v16h A = ld_frag(a0, a0 + 16);
                const _Float16* b0 = &vT[(dh * 16 + l16) * VP + s * 32 + hi * 8];
                v16h Bf = ld_frag(b0, b0 + 16);
                acc = wmma_f16(A, Bf, acc);
            }
            _Float16* os = ks;   // 64 x 32, pitch QP
            #pragma unroll
            for (int g = 0; g < 8; ++g)
                os[(m0 + hi * 8 + g) * QP + dh * 16 + l16] = (_Float16)acc[g];
        }
        __syncthreads();

        // coalesced b128 store of this head's 64x32 output slice
        {
            int row   = tid >> 2;       // 0..63
            int chunk = tid & 3;        // 4 x 8 halves = 32 cols
            uint4 v = *(const uint4*)&ks[row * QP + chunk * 8];
            *(uint4*)&attn_out[(mbase + row) * DIM + head * HD + chunk * 8] = v;
        }
    }
}

// ---------------------------------------------------------------------------
// Kernel 2: output projection (M=262144, N=192, K=192) + bias + inverse roll.
// One block = 128 pixels (2 windows). 256 threads = 8 waves.
// A tile staged via GLOBAL_LOAD_ASYNC_TO_LDS_B128 (ASYNCcnt path).
// ---------------------------------------------------------------------------
__global__ void __launch_bounds__(256)
wa_proj(const _Float16* __restrict__ attn,   // (262144,192) window-major f16
        const _Float16* __restrict__ wo,     // (192,192) f16
        const float* __restrict__ bo,        // (192,)
        float* __restrict__ out)             // (4,256,256,192)
{
    __shared__ _Float16 At[128 * XP];        // 51200 B

    const int tid  = threadIdx.x;
    const int lane = tid & 31;
    const int wv   = tid >> 5;
    const int l16  = lane & 15;
    const int hi   = lane >> 4;

    const long mbase = (long)blockIdx.x * 128;

    // async-stage A tile: 128 rows x 192 halves = 3072 x b128
    #pragma unroll
    for (int it = 0; it < 12; ++it) {
        int k = tid + it * 256;
        int row = k / 24, c16 = k % 24;
        async_copy_b128(&At[row * XP + c16 * 8],
                        attn + (mbase + row) * DIM + c16 * 8);
    }
    async_wait0();
    __syncthreads();

    const int m0 = wv * 16;
    for (int nt = 0; nt < 12; ++nt) {
        int o0 = nt * 16;
        v8f acc;
        #pragma unroll
        for (int z = 0; z < 8; ++z) acc[z] = 0.f;
        #pragma unroll
        for (int s = 0; s < 6; ++s) {
            const _Float16* a0 = &At[(m0 + l16) * XP + s * 32 + hi * 8];
            v16h A = ld_frag(a0, a0 + 16);
            const _Float16* b0 = wo + (o0 + l16) * DIM + s * 32 + hi * 8;
            v16h Bf = ld_frag(b0, b0 + 16);
            acc = wmma_f16(A, Bf, acc);
        }
        int o = o0 + l16;
        float bias = bo[o];

        // rows m0+hi*8 .. +7 share one window/row; only the column rolls.
        long mrow = mbase + m0 + hi * 8;
        int bb2 = (int)(mrow >> 16);
        int wn  = (int)(mrow >> 6) & 1023;
        int pp  = (int)mrow & 63;
        int r   = (((wn >> 5) << 3) + (pp >> 3) + DISP) & 255;
        int c0  = ((wn & 31) << 3) + DISP;
        const long base = (((long)bb2 * 256 + r) * 256) * DIM + o;
        #pragma unroll
        for (int g = 0; g < 8; ++g) {
            int c = (c0 + g) & 255;
            out[base + (long)c * DIM] = acc[g] + bias;
        }
    }
}

// ---------------------------------------------------------------------------
extern "C" void kernel_launch(void* const* d_in, const int* in_sizes, int n_in,
                              void* d_out, int out_size, void* d_ws, size_t ws_size,
                              hipStream_t stream) {
    const float* x     = (const float*)d_in[0];
    const float* wqkv  = (const float*)d_in[1];
    const float* pos   = (const float*)d_in[2];
    const float* wout  = (const float*)d_in[3];
    const float* bout  = (const float*)d_in[4];

    _Float16* wbuf = (_Float16*)d_ws;
    _Float16* wq   = wbuf;                       // 110592 halves
    _Float16* wo   = wbuf + 3 * DIM * DIM;       //  36864 halves
    _Float16* attn = wbuf + 4 * DIM * DIM;       // 262144*192 halves (~96 MB)

    wa_convert_weights<<<576, 256, 0, stream>>>(wqkv, wout, wbuf);
    wa_window_attn<<<4096, 256, 0, stream>>>(x, wq, pos, attn);
    wa_proj<<<2048, 256, 0, stream>>>(attn, wo, bout, (float*)d_out);
}
